// Corrector_37598143710091
// MI455X (gfx1250) — matrix operand order repaired and checked by reading, not compiled, not run on hardware
//
#include <hip/hip_runtime.h>
#include <math.h>

// ---------------------------------------------------------------------------
// Sizes (fixed by the reference)
// ---------------------------------------------------------------------------
#define BB   64
#define NN   4096
#define DIN  192
#define SS   11
#define EE   128
#define HH   256
#define NPAD 200   // LDS row pitch (bf16 elems) for the LN'd x tile
#define NCH  8     // N-chunks for the updates reduction
#define NPC  (NN / NCH)   // 512 n per chunk
#define KSC  (NPC / 32)   // 16 k-steps per chunk

typedef __attribute__((ext_vector_type(16))) __bf16 v16bf;
typedef __attribute__((ext_vector_type(8)))  float  v8f;

union BFrag { v16bf v; uint4 q[2]; unsigned short h[16]; };
union F16u  { float4 f4[4]; float f[16]; };

__device__ inline unsigned short f2bfu(float f) {
  __bf16 h = (__bf16)f;                       // native v_cvt_*bf16_f32
  return __builtin_bit_cast(unsigned short, h);
}
__device__ inline float bf2f(unsigned short h) {
  unsigned u = ((unsigned)h) << 16;
  return __builtin_bit_cast(float, u);
}
__device__ inline v8f wmma_bf16(v16bf a, v16bf b, v8f c) {
  // (neg_a, A, neg_b, B, c_mod, C, reuse_a, reuse_b)
  return __builtin_amdgcn_wmma_f32_16x16x32_bf16(false, a, false, b, (short)0, c,
                                                 false, false);
}
__device__ inline float blockSum128(float v, float* scr) {
  for (int m = 16; m > 0; m >>= 1) v += __shfl_xor(v, m, 32);
  int w = threadIdx.x >> 5;
  if ((threadIdx.x & 31) == 0) scr[w] = v;
  __syncthreads();
  float r = scr[0] + scr[1] + scr[2] + scr[3];
  __syncthreads();
  return r;
}

// ---------------------------------------------------------------------------
// K1: fused LN(inputs) + k1 = x@Wk^T, v1 = x@Wv^T   (bf16 WMMA, f32 accum)
// Block: 128 threads (4 waves), 32 input rows of one batch.
// k1 -> row-major bf16 [b][n][e];  v1 -> B-fragment-packed bf16 for K5.
// ---------------------------------------------------------------------------
__global__ __launch_bounds__(128)
void k_ln_kv(const float* __restrict__ x, const float* __restrict__ g,
             const float* __restrict__ bbias, const float* __restrict__ Wk,
             const float* __restrict__ Wv, unsigned short* __restrict__ k1,
             unsigned short* __restrict__ v1p) {
  __shared__ unsigned short xl[32 * NPAD];
  __shared__ unsigned short stg[4][512];
  const int t = threadIdx.x, lane = t & 31, w = t >> 5;
  const int b = blockIdx.y, nt32 = blockIdx.x;
  const int n0 = nt32 * 32;

  // ---- LayerNorm: 4 threads per row, 48 cols each ----
  {
    const int row = t >> 2, sub = t & 3, c0 = sub * 48;
    const float4* xr4 =
        (const float4*)(x + ((size_t)(b * NN + n0 + row)) * DIN + c0);
    float4 xv[12];
#pragma unroll
    for (int i = 0; i < 12; i++) xv[i] = xr4[i];
    const float* xf = (const float*)xv;
    float s = 0.f, sq = 0.f;
#pragma unroll
    for (int i = 0; i < 48; i++) { float f = xf[i]; s += f; sq += f * f; }
    s  += __shfl_xor(s, 1, 32);  s  += __shfl_xor(s, 2, 32);
    sq += __shfl_xor(sq, 1, 32); sq += __shfl_xor(sq, 2, 32);
    const float mean = s * (1.0f / DIN);
    const float var  = sq * (1.0f / DIN) - mean * mean;
    const float inv  = rsqrtf(var + 1e-5f);
#pragma unroll
    for (int i = 0; i < 48; i++) {
      int c = c0 + i;
      float v = (xf[i] - mean) * inv * g[c] + bbias[c];
      xl[row * NPAD + c] = f2bfu(v);
    }
  }
  __syncthreads();

  // ---- GEMM phase: 16 jobs = {k1,v1} x 8 e-tiles, 4 per wave ----
  const int rg = lane >> 4, lm = lane & 15;
  for (int jj = 0; jj < 4; jj++) {
    const int j = w + 4 * jj;
    const int mat = j >> 3, et = j & 7;
    const float* W = mat ? Wv : Wk;
    v8f acc0, acc1;
#pragma unroll
    for (int v = 0; v < 8; v++) { acc0[v] = 0.f; acc1[v] = 0.f; }
#pragma unroll
    for (int kk = 0; kk < 6; kk++) {
      const int kb = kk * 32;
      BFrag a0, a1, bf;
      a0.q[0] = *(const uint4*)(xl + lm * NPAD + kb + rg * 8);
      a0.q[1] = *(const uint4*)(xl + lm * NPAD + kb + 16 + rg * 8);
      a1.q[0] = *(const uint4*)(xl + (16 + lm) * NPAD + kb + rg * 8);
      a1.q[1] = *(const uint4*)(xl + (16 + lm) * NPAD + kb + 16 + rg * 8);
      // B[k][e] = W[e][k] : 2x8 f32 (k = kb+8rg.., kb+16+8rg..) from row (et*16+lm)
      F16u wu;
      const float4* wr4 =
          (const float4*)(W + (et * 16 + lm) * DIN + kb + rg * 8);
#pragma unroll
      for (int i = 0; i < 4; i++) wu.f4[i] = wr4[i + (i & 2)];
#pragma unroll
      for (int i = 0; i < 16; i++) bf.h[i] = f2bfu(wu.f[i]);
      acc0 = wmma_bf16(a0.v, bf.v, acc0);
      acc1 = wmma_bf16(a1.v, bf.v, acc1);
    }
    if (mat == 0) {  // k1 row-major bf16
      size_t base = ((size_t)b * NN + n0) * EE + et * 16 + lm;
#pragma unroll
      for (int v = 0; v < 8; v++) {
        k1[base + (size_t)(v + 8 * rg) * EE]      = f2bfu(acc0[v]);
        k1[base + (size_t)(16 + v + 8 * rg) * EE] = f2bfu(acc1[v]);
      }
    } else {         // v1: stage [16e][32n] in LDS, emit B-frag pack
      unsigned short* sg = stg[w];
#pragma unroll
      for (int v = 0; v < 8; v++) {
        sg[lm * 32 + v + 8 * rg]      = f2bfu(acc0[v]);
        sg[lm * 32 + 16 + v + 8 * rg] = f2bfu(acc1[v]);
      }
      // same-wave DS ops are in-order; read back in fragment order
      const uint4* pr = (const uint4*)(sg + lm * 32 + rg * 8);
      uint4 q0 = pr[0], q1 = pr[2];
      uint4* dst = (uint4*)(v1p +
          ((((size_t)b * 128 + nt32) * 8 + et) * 32 + lane) * 16);
      dst[0] = q0; dst[1] = q1;
    }
  }
}

// ---------------------------------------------------------------------------
// K2: slots preprocessing (LN + optional MHA + MLP), one block per batch
// ---------------------------------------------------------------------------
__global__ __launch_bounds__(128)
void k_slots_pre(const float* __restrict__ islots, const int* __restrict__ ip,
                 const float* __restrict__ lsg, const float* __restrict__ lsb,
                 const float* __restrict__ inw, const float* __restrict__ inb,
                 const float* __restrict__ ow,  const float* __restrict__ obias,
                 const float* __restrict__ aw1, const float* __restrict__ ab1,
                 const float* __restrict__ aw2, const float* __restrict__ ab2,
                 float* __restrict__ out) {
  const int b = blockIdx.x, t = threadIdx.x;
  __shared__ float s0[SS * EE], qb[SS * EE], kb[SS * EE], vb[SS * EE];
  __shared__ float obuf[SS * EE], scb[4 * SS * SS], hh[HH], scr[4];

  for (int s = 0; s < SS; s++) {
    float xv = islots[((size_t)b * SS + s) * EE + t];
    float mean = blockSum128(xv, scr) * (1.0f / EE);
    float d = xv - mean;
    float var = blockSum128(d * d, scr) * (1.0f / EE);
    s0[s * EE + t] = d * rsqrtf(var + 1e-5f) * lsg[t] + lsb[t];
  }
  __syncthreads();

  const int iv = *ip;
  if (iv == 1) {
    for (int s = 0; s < SS; s++)
      out[((size_t)b * SS + s) * EE + t] = s0[s * EE + t];
    return;
  }

  // qkv projection
  for (int s = 0; s < SS; s++)
    for (int j3 = 0; j3 < 3; j3++) {
      int row = j3 * EE + t;
      float a = inb[row];
      for (int k = 0; k < EE; k++) a += s0[s * EE + k] * inw[row * EE + k];
      (j3 == 0 ? qb : j3 == 1 ? kb : vb)[s * EE + t] = a;
    }
  __syncthreads();

  // scores (4 heads, hd=32), scale 1/sqrt(32)
  for (int p = t; p < 4 * SS * SS; p += 128) {
    int h = p / (SS * SS), r = p % (SS * SS), qi = r / SS, kj = r % SS;
    float a = 0.f;
    for (int d = 0; d < 32; d++)
      a += qb[qi * EE + h * 32 + d] * kb[kj * EE + h * 32 + d];
    scb[p] = a * 0.17677669529663689f;
  }
  __syncthreads();
  for (int p = t; p < 4 * SS; p += 128) {
    int h = p / SS, qi = p % SS, base = h * SS * SS + qi * SS;
    float mx = -3e38f;
    for (int j = 0; j < SS; j++) mx = fmaxf(mx, scb[base + j]);
    float sm = 0.f;
    for (int j = 0; j < SS; j++) { float e = __expf(scb[base + j] - mx); scb[base + j] = e; sm += e; }
    float inv = 1.f / sm;
    for (int j = 0; j < SS; j++) scb[base + j] *= inv;
  }
  __syncthreads();
  {
    int h = t >> 5;
    for (int qi = 0; qi < SS; qi++) {
      float a = 0.f;
      for (int kj = 0; kj < SS; kj++)
        a += scb[h * SS * SS + qi * SS + kj] * vb[kj * EE + t];
      obuf[qi * EE + t] = a;
    }
  }
  __syncthreads();

  // out-proj + residual + LN + MLP, per slot
  for (int s = 0; s < SS; s++) {
    float a = obias[t];
    for (int k = 0; k < EE; k++) a += obuf[s * EE + k] * ow[t * EE + k];
    float s1v = s0[s * EE + t] + a;
    float mean = blockSum128(s1v, scr) * (1.0f / EE);
    float d = s1v - mean;
    float var = blockSum128(d * d, scr) * (1.0f / EE);
    float s2v = d * rsqrtf(var + 1e-5f) * lsg[t] + lsb[t];
    s0[s * EE + t] = s2v;
    __syncthreads();
    for (int kk = t; kk < HH; kk += 128) {
      float hv = ab1[kk];
      for (int k = 0; k < EE; k++) hv += s0[s * EE + k] * aw1[kk * EE + k];
      hh[kk] = fmaxf(hv, 0.f);
    }
    __syncthreads();
    float o2 = ab2[t];
    for (int k = 0; k < HH; k++) o2 += hh[k] * aw2[t * HH + k];
    out[((size_t)b * SS + s) * EE + t] = s2v + o2;
    __syncthreads();
  }
}

// ---------------------------------------------------------------------------
// K3: q = LN(slots)@Wq^T -> bf16 qbuf [b][16][128] (rows 11..15 zeroed)
// ---------------------------------------------------------------------------
__global__ __launch_bounds__(128)
void k_q(const float* __restrict__ slots, const float* __restrict__ lsg,
         const float* __restrict__ lsb, const float* __restrict__ Wq,
         unsigned short* __restrict__ qbuf) {
  const int b = blockIdx.x, t = threadIdx.x;
  __shared__ float sn[EE], scr[4];
  for (int s = 0; s < SS; s++) {
    float xv = slots[((size_t)b * SS + s) * EE + t];
    float mean = blockSum128(xv, scr) * (1.0f / EE);
    float d = xv - mean;
    float var = blockSum128(d * d, scr) * (1.0f / EE);
    sn[t] = d * rsqrtf(var + 1e-5f) * lsg[t] + lsb[t];
    __syncthreads();
    float q = 0.f;
    for (int k = 0; k < EE; k++) q += sn[k] * Wq[t * EE + k];
    qbuf[((size_t)b * 16 + s) * EE + t] = f2bfu(q);
    __syncthreads();
  }
  for (int s = SS; s < 16; s++) qbuf[((size_t)b * 16 + s) * EE + t] = 0;
}

// ---------------------------------------------------------------------------
// K4: logits = scale * q @ k1^T  (WMMA), softmax over s, +eps, -> attnT bf16
// One wave per (b, 16-row n-tile). D: lane holds col n=L%16, rows s=v+8*(L/16).
// ---------------------------------------------------------------------------
__global__ __launch_bounds__(32)
void k_attn(const unsigned short* __restrict__ k1,
            const unsigned short* __restrict__ qbuf,
            unsigned short* __restrict__ attnT) {
  const int lane = threadIdx.x, nt = blockIdx.x, b = blockIdx.y;
  const int n0 = nt * 16, lm = lane & 15, rg = lane >> 4;
  v8f acc;
#pragma unroll
  for (int v = 0; v < 8; v++) acc[v] = 0.f;
#pragma unroll
  for (int kk = 0; kk < 4; kk++) {
    const int kb = kk * 32;
    BFrag a, bf;
    const unsigned short* qp = qbuf + ((size_t)b * 16 + lm) * EE + kb + rg * 8;
    a.q[0] = *(const uint4*)qp;
    a.q[1] = *(const uint4*)(qp + 16);
    const uint4* kp =
        (const uint4*)(k1 + ((size_t)b * NN + n0 + lm) * EE + kb + rg * 8);
    bf.q[0] = kp[0]; bf.q[1] = kp[2];
    acc = wmma_bf16(a.v, bf.v, acc);
  }
  const float sc = 0.08838834764831845f;           // 128^-0.5
  const int nrel = rg ? 3 : 8;                     // valid s rows in this half
  float mloc = -3e38f;
#pragma unroll
  for (int v = 0; v < 8; v++)
    if (v < nrel) mloc = fmaxf(mloc, acc[v] * sc);
  float mm = fmaxf(mloc, __shfl_xor(mloc, 16, 32));
  float p[8]; float ss = 0.f;
#pragma unroll
  for (int v = 0; v < 8; v++) {
    p[v] = (v < nrel) ? __expf(acc[v] * sc - mm) : 0.f;
    ss += p[v];
  }
  float tot = ss + __shfl_xor(ss, 16, 32);
  float inv = 1.f / tot;
#pragma unroll
  for (int v = 0; v < 8; v++) {
    if (v < nrel) {
      int s = v + 8 * rg;
      float av = p[v] * inv + 1e-8f;
      attnT[((size_t)b * 16 + s) * NN + n0 + lm] = f2bfu(av);
    }
  }
}

// ---------------------------------------------------------------------------
// K5: partial colsum + partial updates = attn^T @ v1 over one N-chunk (WMMA)
// Grid (NCH, B), 256 threads (8 waves, wave = e-tile), 16 k-steps per chunk.
// Partials are reduced (deterministic fixed order) in K6.
// ---------------------------------------------------------------------------
__global__ __launch_bounds__(256)
void k_updates_part(const unsigned short* __restrict__ attnT,
                    const unsigned short* __restrict__ v1p,
                    float* __restrict__ pupd, float* __restrict__ pcs) {
  __shared__ float scr[8];
  const int ch = blockIdx.x, b = blockIdx.y, t = threadIdx.x;
  const int w = t >> 5, lane = t & 31, lm = lane & 15, rg = lane >> 4;

  // partial column sums over this chunk's 512 n values
  for (int s = 0; s < SS; s++) {
    float sum = 0.f;
    for (int n = t; n < NPC; n += 256)
      sum += bf2f(attnT[((size_t)b * 16 + s) * NN + ch * NPC + n]);
    for (int m = 16; m > 0; m >>= 1) sum += __shfl_xor(sum, m, 32);
    if (lane == 0) scr[w] = sum;
    __syncthreads();
    if (t == 0) {
      float tt = 0.f;
      for (int i = 0; i < 8; i++) tt += scr[i];
      pcs[((size_t)b * NCH + ch) * 16 + s] = tt;
    }
    __syncthreads();
  }

  // partial GEMM over 16 k-steps of 32
  v8f acc;
#pragma unroll
  for (int v = 0; v < 8; v++) acc[v] = 0.f;
  for (int ks = 0; ks < KSC; ks++) {
    const int ksg = ch * KSC + ks;
    const int nb = ksg * 32;
    BFrag a, bf;
    const unsigned short* ap =
        attnT + ((size_t)b * 16 + lm) * NN + nb + rg * 8;
    a.q[0] = *(const uint4*)ap;
    a.q[1] = *(const uint4*)(ap + 16);
    const uint4* bp = (const uint4*)(v1p +
        ((((size_t)b * 128 + ksg) * 8 + w) * 32 + lane) * 16);
    bf.q[0] = bp[0]; bf.q[1] = bp[1];
    if (ks + 2 < KSC) {  // cover L2 latency on the streamed fragments
      __builtin_prefetch((const void*)(v1p +
          ((((size_t)b * 128 + ksg + 2) * 8 + w) * 32 + lane) * 16), 0, 1);
    }
    acc = wmma_bf16(a.v, bf.v, acc);
  }
#pragma unroll
  for (int v = 0; v < 8; v++) {
    int s = v + 8 * rg;
    if (s < SS)
      pupd[(((size_t)b * NCH + ch) * 16 + s) * EE + w * 16 + lm] = acc[v];
  }
}

// ---------------------------------------------------------------------------
// K6: chunk reduction + renorm + GRU cell + LN + final MLP
// One block per (slot, batch)
// ---------------------------------------------------------------------------
__global__ __launch_bounds__(128)
void k_gru_mlp(const float* __restrict__ pupd, const float* __restrict__ pcs,
               const float* __restrict__ wih,
               const float* __restrict__ whh, const float* __restrict__ bih,
               const float* __restrict__ bhh, const float* __restrict__ lmg,
               const float* __restrict__ lmb, const float* __restrict__ w1,
               const float* __restrict__ b1, const float* __restrict__ w2,
               const float* __restrict__ b2, float* __restrict__ out) {
  const int s = blockIdx.x, b = blockIdx.y, t = threadIdx.x;
  __shared__ float up[EE], hp[EE], mmv[EE], hh[HH], scr[4];
  float cs = 0.f, u = 0.f;
#pragma unroll
  for (int c = 0; c < NCH; c++) {
    cs += pcs[((size_t)b * NCH + c) * 16 + s];
    u  += pupd[(((size_t)b * NCH + c) * 16 + s) * EE + t];
  }
  u *= (1.f / cs);
  float h = out[((size_t)b * SS + s) * EE + t];
  up[t] = u; hp[t] = h;
  __syncthreads();
  float gir = bih[t], giz = bih[EE + t], gin = bih[2 * EE + t];
  float ghr = bhh[t], ghz = bhh[EE + t], ghn = bhh[2 * EE + t];
  for (int k = 0; k < EE; k++) {
    float uk = up[k], hk = hp[k];
    gir += uk * wih[t * EE + k];
    giz += uk * wih[(EE + t) * EE + k];
    gin += uk * wih[(2 * EE + t) * EE + k];
    ghr += hk * whh[t * EE + k];
    ghz += hk * whh[(EE + t) * EE + k];
    ghn += hk * whh[(2 * EE + t) * EE + k];
  }
  float r = 1.f / (1.f + __expf(-(gir + ghr)));
  float z = 1.f / (1.f + __expf(-(giz + ghz)));
  float n = tanhf(gin + r * ghn);
  float hnew = (1.f - z) * n + z * h;
  float mean = blockSum128(hnew, scr) * (1.0f / EE);
  float d = hnew - mean;
  float var = blockSum128(d * d, scr) * (1.0f / EE);
  float mv = d * rsqrtf(var + 1e-5f) * lmg[t] + lmb[t];
  mmv[t] = mv;
  __syncthreads();
  for (int kk = t; kk < HH; kk += 128) {
    float a = b1[kk];
    for (int k = 0; k < EE; k++) a += mmv[k] * w1[kk * EE + k];
    hh[kk] = fmaxf(a, 0.f);
  }
  __syncthreads();
  float o = b2[t];
  for (int k = 0; k < HH; k++) o += hh[k] * w2[t * HH + k];
  out[((size_t)b * SS + s) * EE + t] = hnew + o;
}

// ---------------------------------------------------------------------------
// Launch
// ---------------------------------------------------------------------------
extern "C" void kernel_launch(void* const* d_in, const int* in_sizes, int n_in,
                              void* d_out, int out_size, void* d_ws,
                              size_t ws_size, hipStream_t stream) {
  (void)in_sizes; (void)n_in; (void)out_size; (void)ws_size;
  const float* inputs  = (const float*)d_in[0];
  const float* islots  = (const float*)d_in[1];
  const int*   ip      = (const int*)d_in[2];
  const float* ln_in_g = (const float*)d_in[3];
  const float* ln_in_b = (const float*)d_in[4];
  const float* ln_s_g  = (const float*)d_in[5];
  const float* ln_s_b  = (const float*)d_in[6];
  const float* ln_m_g  = (const float*)d_in[7];
  const float* ln_m_b  = (const float*)d_in[8];
  const float* mha_in_w  = (const float*)d_in[9];
  const float* mha_in_b  = (const float*)d_in[10];
  const float* mha_out_w = (const float*)d_in[11];
  const float* mha_out_b = (const float*)d_in[12];
  const float* attn_w1 = (const float*)d_in[13];
  const float* attn_b1 = (const float*)d_in[14];
  const float* attn_w2 = (const float*)d_in[15];
  const float* attn_b2 = (const float*)d_in[16];
  const float* Wq = (const float*)d_in[17];
  const float* Wk = (const float*)d_in[18];
  const float* Wv = (const float*)d_in[19];
  const float* gru_wih = (const float*)d_in[20];
  const float* gru_whh = (const float*)d_in[21];
  const float* gru_bih = (const float*)d_in[22];
  const float* gru_bhh = (const float*)d_in[23];
  const float* mlp_w1 = (const float*)d_in[24];
  const float* mlp_b1 = (const float*)d_in[25];
  const float* mlp_w2 = (const float*)d_in[26];
  const float* mlp_b2 = (const float*)d_in[27];
  float* out = (float*)d_out;

  char* ws = (char*)d_ws;
  const size_t SZ_K1   = (size_t)BB * NN * EE * 2;        // 67 MB bf16
  const size_t SZ_V1P  = (size_t)BB * NN * EE * 2;        // 67 MB bf16 pack
  const size_t SZ_ATT  = (size_t)BB * 16 * NN * 2;        // 8.4 MB bf16
  const size_t SZ_Q    = (size_t)BB * 16 * EE * 2;        // 256 KB bf16
  const size_t SZ_PUPD = (size_t)BB * NCH * 16 * EE * 4;  // 4 MB f32
  unsigned short* k1    = (unsigned short*)(ws);
  unsigned short* v1p   = (unsigned short*)(ws + SZ_K1);
  unsigned short* attnT = (unsigned short*)(ws + SZ_K1 + SZ_V1P);
  unsigned short* qbuf  = (unsigned short*)(ws + SZ_K1 + SZ_V1P + SZ_ATT);
  float* pupd = (float*)(ws + SZ_K1 + SZ_V1P + SZ_ATT + SZ_Q);
  float* pcs  = (float*)(ws + SZ_K1 + SZ_V1P + SZ_ATT + SZ_Q + SZ_PUPD);

  k_ln_kv<<<dim3(NN / 32, BB), 128, 0, stream>>>(inputs, ln_in_g, ln_in_b, Wk,
                                                 Wv, k1, v1p);
  k_slots_pre<<<BB, 128, 0, stream>>>(islots, ip, ln_s_g, ln_s_b, mha_in_w,
                                      mha_in_b, mha_out_w, mha_out_b, attn_w1,
                                      attn_b1, attn_w2, attn_b2, out);
  for (int it = 0; it < 2; it++) {
    k_q<<<BB, 128, 0, stream>>>(out, ln_s_g, ln_s_b, Wq, qbuf);
    k_attn<<<dim3(NN / 16, BB), 32, 0, stream>>>(k1, qbuf, attnT);
    k_updates_part<<<dim3(NCH, BB), 256, 0, stream>>>(attnT, v1p, pupd, pcs);
    k_gru_mlp<<<dim3(SS, BB), 128, 0, stream>>>(pupd, pcs, gru_wih, gru_whh,
                                                gru_bih, gru_bhh, ln_m_g,
                                                ln_m_b, mlp_w1, mlp_b1, mlp_w2,
                                                mlp_b2, out);
  }
}
